// HeteroGraphEncoder_26422638805246
// MI455X (gfx1250) — matrix operand order repaired and checked by reading, not compiled, not run on hardware
//
#include <hip/hip_runtime.h>

typedef __attribute__((ext_vector_type(16))) _Float16 v16h;
typedef __attribute__((ext_vector_type(8)))  float    v8f;

#define NBn 30000
#define NGn 6000
#define NBp 30208   // padded bus rows: multiple of 256 (8 waves x 32 rows per block)
#define NGp 6144    // padded gen rows
#define ELn 250000
#define EGn 12000
#define HCn 256

// ---------- helpers ----------
static __device__ __forceinline__ unsigned enc_f32(float f) {
  unsigned u = __float_as_uint(f);
  return (u >> 31) ? ~u : (u | 0x80000000u);
}
static __device__ __forceinline__ float dec_f32(unsigned e) {
  unsigned u = (e >> 31) ? (e ^ 0x80000000u) : ~e;
  return __uint_as_float(u);
}

__global__ void k_fill_f32(float* p, float v, int n) {
  int i = blockIdx.x * blockDim.x + threadIdx.x;
  if (i < n) p[i] = v;
}
__global__ void k_fill_u32(unsigned* p, unsigned v, int n) {
  int i = blockIdx.x * blockDim.x + threadIdx.x;
  if (i < n) p[i] = v;
}
__global__ void k_f32_to_f16(const float* __restrict__ s, _Float16* __restrict__ d, int n) {
  int i = blockIdx.x * blockDim.x + threadIdx.x;
  if (i < n) d[i] = (_Float16)s[i];
}
// W: K x N (f32) -> Wt: N x K (f16)
__global__ void k_transpose_f16(const float* __restrict__ W, _Float16* __restrict__ Wt,
                                int K, int N) {
  int i = blockIdx.x * blockDim.x + threadIdx.x;
  if (i >= K * N) return;
  int k = i / N, n = i % N;
  Wt[n * K + k] = (_Float16)W[i];
}
// gen projector: 6000x16 @ 16x32 + b
__global__ void k_proj_gen(const float* __restrict__ xg, const float* __restrict__ w,
                           const float* __restrict__ b, float* __restrict__ o) {
  int i = blockIdx.x * blockDim.x + threadIdx.x;
  if (i >= NGn * 32) return;
  int n = i >> 5, j = i & 31;
  float acc = b[j];
#pragma unroll
  for (int k = 0; k < 16; k++) acc += xg[n * 16 + k] * w[k * 32 + j];
  o[i] = acc;
}

// ---------- WMMA GEMM: C(MxN,f32) = A(MxK,f16) * Bt^T (Bt is NxK,f16) ----------
// M is pre-padded to a multiple of 256 -> grid tiles exactly, no tails, no exec masking.
// block = 256 threads = 8 waves; B strip (64 cols x K) staged once in LDS for all waves.
// Each wave owns a 32x64 tile (2 A-frags, 8 accumulators).  K-loop is unrolled by two
// with even/odd fragment registers so A prefetch overlaps the WMMA groups without moves.
static __device__ __forceinline__ void load_afrag(v16h& x, const _Float16* __restrict__ p) {
#pragma unroll
  for (int i = 0; i < 8; i++) { x[i] = p[i]; x[8 + i] = p[8 + i]; }
}

__global__ void __launch_bounds__(256) k_gemm(
    const _Float16* __restrict__ A, int lda,
    const _Float16* __restrict__ Bt,
    float* __restrict__ C, int ldc,
    int N, int K,
    const float* __restrict__ bias) {
  __shared__ _Float16 Bs[64 * 256];          // 32 KB max (K <= 256)
  int wave = threadIdx.x >> 5;
  int lane = threadIdx.x & 31;
  int tileN = blockIdx.y * 64;

  { // cooperative contiguous copy of Bt rows [tileN, tileN+64) into LDS
    const uint4* gsrc = (const uint4*)(Bt + (size_t)tileN * K);
    uint4* ldst = (uint4*)Bs;
    int chunks = (64 * K) >> 3;              // 16B chunks
    for (int i = threadIdx.x; i < chunks; i += 256) ldst[i] = gsrc[i];
  }
  __syncthreads();

  int tileM = (blockIdx.x * 8 + wave) * 32;
  int hh = lane >> 4, lr = lane & 15;

  v8f acc[8];
#pragma unroll
  for (int j = 0; j < 8; j++) acc[j] = (v8f){};

  const _Float16* ap0 = A + (size_t)(tileM + lr) * lda + hh * 16;
  const _Float16* ap1 = ap0 + (size_t)16 * lda;
  const _Float16* bs = Bs + (size_t)lr * K + hh * 16;

  v16h e0, e1, o0, o1;
  load_afrag(e0, ap0);
  load_afrag(e1, ap1);

  for (int kk = 0; kk < K; kk += 64) {
    if (kk + 32 < K) {                       // prefetch odd step
      load_afrag(o0, ap0 + kk + 32);
      load_afrag(o1, ap1 + kk + 32);
    }
#pragma unroll
    for (int j = 0; j < 4; j++) {
      const v16h b = *(const v16h*)(bs + (size_t)j * 16 * K + kk);
      acc[j]     = __builtin_amdgcn_wmma_f32_16x16x32_f16(false, e0, false, b, (short)0, acc[j],     false, false);
      acc[4 + j] = __builtin_amdgcn_wmma_f32_16x16x32_f16(false, e1, false, b, (short)0, acc[4 + j], false, false);
    }
    if (kk + 64 < K) {                       // prefetch next even step
      load_afrag(e0, ap0 + kk + 64);
      load_afrag(e1, ap1 + kk + 64);
    }
    if (kk + 32 < K) {
#pragma unroll
      for (int j = 0; j < 4; j++) {
        const v16h b = *(const v16h*)(bs + (size_t)j * 16 * K + kk + 32);
        acc[j]     = __builtin_amdgcn_wmma_f32_16x16x32_f16(false, o0, false, b, (short)0, acc[j],     false, false);
        acc[4 + j] = __builtin_amdgcn_wmma_f32_16x16x32_f16(false, o1, false, b, (short)0, acc[4 + j], false, false);
      }
    }
  }

#pragma unroll
  for (int j = 0; j < 4; j++) {
    int col = tileN + j * 16 + lr;
    float bv = bias ? bias[col] : 0.f;
#pragma unroll
    for (int r = 0; r < 8; r++) {
      int row = tileM + r + hh * 8;
      C[(size_t)row * ldc + col] = acc[j][r] + bv;
      C[(size_t)(row + 16) * ldc + col] = acc[4 + j][r] + bv;
    }
  }
}

// ---------- edge kernels (wave per edge, 8 channels per lane) ----------
__global__ void k_edge_logits(const float* __restrict__ xl, const float* __restrict__ xr,
                              const int* __restrict__ src, const int* __restrict__ dst,
                              const float* __restrict__ ea, const float* __restrict__ We,
                              const float* __restrict__ att, const float* __restrict__ temp_p,
                              float* __restrict__ logits, unsigned* __restrict__ mxenc, int E) {
  int wave = threadIdx.x >> 5, lane = threadIdx.x & 31;
  int e = blockIdx.x * 8 + wave;
  if (e >= E) return;
  int s = src[e], d = dst[e];
  float a6[6];
#pragma unroll
  for (int j = 0; j < 6; j++) a6[j] = ea[e * 6 + j];
  a6[5] = temp_p[0] / fmaxf(a6[3], 1e-6f);
  int c0 = lane * 8;
  const float* xls = xl + (size_t)s * HCn + c0;
  const float* xrd = xr + (size_t)d * HCn + c0;
  float part = 0.f;
#pragma unroll
  for (int i = 0; i < 8; i++) {
    int c = c0 + i;
    float m = xls[i] + xrd[i];
#pragma unroll
    for (int j = 0; j < 6; j++) m += a6[j] * We[j * HCn + c];
    m = m > 0.f ? m : 0.2f * m;      // leaky_relu(0.2)
    part += m * att[c];
  }
  part += __shfl_xor(part, 1, 32);
  part += __shfl_xor(part, 2, 32);
  part += __shfl_xor(part, 4, 32);
  if ((lane & 7) == 0) {
    int h = lane >> 3;
    logits[e * 4 + h] = part;
    atomicMax(&mxenc[(size_t)d * 4 + h], enc_f32(part));
  }
}

__global__ void k_edge_exp(float* __restrict__ logits, const int* __restrict__ dst,
                           const unsigned* __restrict__ mxenc, float* __restrict__ sums, int E) {
  int i = blockIdx.x * blockDim.x + threadIdx.x;
  if (i >= E * 4) return;
  int e = i >> 2, h = i & 3;
  int d = dst[e];
  unsigned enc = mxenc[(size_t)d * 4 + h];
  float mx = 0.f;
  if (enc != 0u) {
    float f = dec_f32(enc);
    mx = isfinite(f) ? f : 0.f;
  }
  float ex = expf(logits[i] - mx);
  logits[i] = ex;
  atomicAdd(&sums[(size_t)d * 4 + h], ex);
}

__global__ void k_edge_agg(const float* __restrict__ xl, const float* __restrict__ ex,
                           const int* __restrict__ src, const int* __restrict__ dst,
                           const float* __restrict__ sums, float* __restrict__ O, int E) {
  int wave = threadIdx.x >> 5, lane = threadIdx.x & 31;
  int e = blockIdx.x * 8 + wave;
  if (e >= E) return;
  int s = src[e], d = dst[e];
  int c0 = lane * 8, h = lane >> 3;
  float alpha = ex[e * 4 + h] / (sums[(size_t)d * 4 + h] + 1e-16f);
  const float* xls = xl + (size_t)s * HCn + c0;
  float* Od = O + (size_t)d * HCn + c0;
#pragma unroll
  for (int i = 0; i < 8; i++) atomicAdd(&Od[i], xls[i] * alpha);
}

// ---------- finalize: bias + (head-mean) + LayerNorm + residual + ELU ----------
__global__ void k_finalize(float* __restrict__ out, int ostr,
                           const float* __restrict__ O,
                           const float* __restrict__ ba, const float* __restrict__ bb,
                           const float* __restrict__ lnw, const float* __restrict__ lnb,
                           const float* __restrict__ res, int rstr,
                           int dout, int meanheads, int elu) {
  int n = blockIdx.x, t = threadIdx.x;
  __shared__ float s1[256], s2[256];
  float v = 0.f;
  if (t < dout) {
    const float* r = O + (size_t)n * HCn;
    if (meanheads)
      v = 0.25f * (r[t] + r[64 + t] + r[128 + t] + r[192 + t]);
    else
      v = r[t];
    v += ba[t];
    if (bb) v += bb[t];
  }
  s1[t] = (t < dout) ? v : 0.f;
  s2[t] = (t < dout) ? v * v : 0.f;
  __syncthreads();
  for (int off = 128; off > 0; off >>= 1) {
    if (t < off) { s1[t] += s1[t + off]; s2[t] += s2[t + off]; }
    __syncthreads();
  }
  float mu = s1[0] / dout;
  float var = s2[0] / dout - mu * mu;
  if (t < dout) {
    float y = (v - mu) * rsqrtf(var + 1e-5f) * lnw[t] + lnb[t] + res[(size_t)n * rstr + t];
    if (elu && y < 0.f) y = expf(y) - 1.f;
    out[(size_t)n * ostr + t] = y;
  }
}

// ---------- host orchestration ----------
extern "C" void kernel_launch(void* const* d_in, const int* in_sizes, int n_in,
                              void* d_out, int out_size, void* d_ws, size_t ws_size,
                              hipStream_t stream) {
  const float* x_bus = (const float*)d_in[0];
  const float* x_gen = (const float*)d_in[1];
  const float* ea_line = (const float*)d_in[2];
  const float* ea_g2b = (const float*)d_in[3];
  const float* ea_b2g = (const float*)d_in[4];
  const float* pgw = (const float*)d_in[5];
  const float* pgb = (const float*)d_in[6];
  const float* rp0w = (const float*)d_in[31];
  const float* rp0b = (const float*)d_in[32];
  const float* rp2w = (const float*)d_in[33];
  const float* rp2b = (const float*)d_in[34];
  const int* src_line = (const int*)d_in[35];
  const int* dst_line = (const int*)d_in[36];
  const int* src_g2b = (const int*)d_in[37];
  const int* dst_g2b = (const int*)d_in[38];
  const int* src_b2g = (const int*)d_in[39];
  const int* dst_b2g = (const int*)d_in[40];

  // workspace carve (node buffers padded to NBp/NGp rows so GEMM pad rows stay in-bounds)
  char* wp = (char*)d_ws;
  auto carve = [&](size_t bytes) -> void* {
    void* r = (void*)wp;
    wp += (bytes + 255) & ~(size_t)255;
    return r;
  };
  float*    XG0  = (float*)   carve((size_t)NGp * 32 * 4);
  _Float16* XB0h = (_Float16*)carve((size_t)NBp * 32 * 2);
  _Float16* XG0h = (_Float16*)carve((size_t)NGp * 32 * 2);
  _Float16* XBh  = (_Float16*)carve((size_t)NBp * 256 * 2);
  _Float16* XGh  = (_Float16*)carve((size_t)NGp * 256 * 2);
  float*    XL   = (float*)   carve((size_t)NBp * 256 * 4);
  float*    XR   = (float*)   carve((size_t)NBp * 256 * 4);
  float*    OB   = (float*)   carve((size_t)NBp * 256 * 4);
  float*    OG   = (float*)   carve((size_t)NGp * 256 * 4);
  float*    B1   = (float*)   carve((size_t)NBp * 256 * 4);
  float*    B2   = (float*)   carve((size_t)NBp * 256 * 4);
  float*    G1   = (float*)   carve((size_t)NGp * 256 * 4);
  float*    G2   = (float*)   carve((size_t)NGp * 256 * 4);
  float*    LG   = (float*)   carve((size_t)ELn * 4 * 4);
  unsigned* MX   = (unsigned*)carve((size_t)NBn * 4 * 4);
  float*    SUM  = (float*)   carve((size_t)NBn * 4 * 4);
  _Float16* WT   = (_Float16*)carve((size_t)256 * 256 * 2);

  auto grd = [](int n) { return dim3((n + 255) / 256); };

  // Mp must be NBp or NGp (multiple of 256)
  auto run_gemm = [&](const _Float16* A, int lda, const float* W, int K, int N,
                      float* C, int ldc, int Mp, const float* bias) {
    k_transpose_f16<<<grd(K * N), 256, 0, stream>>>(W, WT, K, N);
    dim3 g(Mp / 256, N / 64);
    k_gemm<<<g, 256, 0, stream>>>(A, lda, WT, C, ldc, N, K, bias);
  };

  auto run_conv = [&](const _Float16* As, int Msp, const _Float16* Ad, int Mdp, int Md,
                      int din,
                      const float* Wl, const float* Wr, const float* We,
                      const float* att, const float* temp_p,
                      const float* ea, const int* src, const int* dst, int E, float* O) {
    run_gemm(As, din, Wl, din, 256, XL, 256, Msp, nullptr);
    run_gemm(Ad, din, Wr, din, 256, XR, 256, Mdp, nullptr);
    k_fill_u32<<<grd(Md * 4), 256, 0, stream>>>(MX, 0u, Md * 4);
    k_fill_f32<<<grd(Md * 4), 256, 0, stream>>>(SUM, 0.f, Md * 4);
    k_edge_logits<<<E / 8, 256, 0, stream>>>(XL, XR, src, dst, ea, We, att, temp_p, LG, MX, E);
    k_edge_exp<<<grd(E * 4), 256, 0, stream>>>(LG, dst, MX, SUM, E);
    k_edge_agg<<<E / 8, 256, 0, stream>>>(XL, LG, src, dst, SUM, O, E);
  };

  // prologue: input conversions + gen projection
  k_f32_to_f16<<<grd(NBn * 32), 256, 0, stream>>>(x_bus, XB0h, NBn * 32);
  k_proj_gen<<<grd(NGn * 32), 256, 0, stream>>>(x_gen, pgw, pgb, XG0);
  k_f32_to_f16<<<grd(NGn * 32), 256, 0, stream>>>(XG0, XG0h, NGn * 32);

  for (int l = 0; l < 3; l++) {
    int din = (l == 0) ? 32 : 256;
    int dout = (l == 2) ? 64 : 256;
    const float* Wl   = (const float*)d_in[7 + 8 * l];
    const float* Wr   = (const float*)d_in[8 + 8 * l];
    const float* We   = (const float*)d_in[9 + 8 * l];
    const float* att  = (const float*)d_in[10 + 8 * l];
    const float* bc   = (const float*)d_in[11 + 8 * l];
    const float* temp = (const float*)d_in[12 + 8 * l];
    const float* lnw  = (const float*)d_in[13 + 8 * l];
    const float* lnb  = (const float*)d_in[14 + 8 * l];
    const _Float16* Ab = (l == 0) ? XB0h : XBh;
    const _Float16* Ag = (l == 0) ? XG0h : XGh;

    k_fill_f32<<<grd(NBn * 256), 256, 0, stream>>>(OB, 0.f, NBn * 256);
    k_fill_f32<<<grd(NGn * 256), 256, 0, stream>>>(OG, 0.f, NGn * 256);

    // line: bus -> bus
    run_conv(Ab, NBp, Ab, NBp, NBn, din, Wl, Wr, We, att, temp,
             ea_line, src_line, dst_line, ELn, OB);
    // g2b: gen -> bus
    run_conv(Ag, NGp, Ab, NBp, NBn, din, Wl + (size_t)din * 256, Wr + (size_t)din * 256,
             We + 6 * 256, att + 256, temp + 1, ea_g2b, src_g2b, dst_g2b, EGn, OB);
    // b2g: bus -> gen
    run_conv(Ab, NBp, Ag, NGp, NGn, din, Wl + (size_t)2 * din * 256, Wr + (size_t)2 * din * 256,
             We + 12 * 256, att + 512, temp + 2, ea_b2g, src_b2g, dst_b2g, EGn, OG);

    float *outB, *outG;
    const float *resB, *resG;
    int ostr, rstr;
    if (l == 0) {
      outB = B1; outG = G1; ostr = 256; rstr = 256;
      run_gemm(XB0h, 32, rp0w, 32, 256, B1, 256, NBp, rp0b);
      run_gemm(XG0h, 32, rp0w + 32 * 256, 32, 256, G1, 256, NGp, rp0b + 256);
      resB = B1; resG = G1;
    } else if (l == 1) {
      outB = B2; outG = G2; ostr = 256; rstr = 256;
      resB = B1; resG = G1;
    } else {
      // residual projections into free scratch (pad rows absorb GEMM over-run),
      // finalize writes exact rows into d_out
      run_gemm(XBh, 256, rp2w, 256, 64, B1, 64, NBp, rp2b);
      run_gemm(XGh, 256, rp2w + 256 * 64, 256, 64, G1, 64, NGp, rp2b + 64);
      outB = (float*)d_out; outG = (float*)d_out + (size_t)NBn * 64;
      ostr = 64; rstr = 64;
      resB = B1; resG = G1;
    }
    int mh = (l == 2) ? 1 : 0;
    int el = (l < 2) ? 1 : 0;
    k_finalize<<<NBn, 256, 0, stream>>>(outB, ostr, OB, bc, bc + dout,
                                        lnw, lnb, resB, rstr, dout, mh, el);
    k_finalize<<<NGn, 256, 0, stream>>>(outG, ostr, OG, bc + 2 * dout, nullptr,
                                        lnw + dout, lnb + dout, resG, rstr, dout, mh, el);
    if (l < 2) {
      k_f32_to_f16<<<grd(NBn * 256), 256, 0, stream>>>(outB, XBh, NBn * 256);
      k_f32_to_f16<<<grd(NGn * 256), 256, 0, stream>>>(outG, XGh, NGn * 256);
    }
  }
  (void)in_sizes; (void)n_in; (void)out_size; (void)ws_size;
}
